// AttentionBranch_64707977282007
// MI455X (gfx1250) — matrix operand order repaired and checked by reading, not compiled, not run on hardware
//
#include <hip/hip_runtime.h>
#include <hip/hip_bf16.h>

// ---------- CDNA5 vector types ----------
typedef __attribute__((ext_vector_type(16))) _Float16 v16h;
typedef __attribute__((ext_vector_type(8)))  _Float16 v8h;
typedef __attribute__((ext_vector_type(8)))  float    v8f;
typedef __attribute__((ext_vector_type(4)))  float    v4f;

union V16 { v16h v; v8h h[2]; };

// =====================================================================
// Weight pack: w (OIHW fp32, flat o*K + k with k = ci*9+kh*3+kw) ->
// Bp[((k/32)*N + n)*32 + (k%32)] (f16): per 32-wide K block, the
// 32-element column segment for output channel n is contiguous.
// =====================================================================
__global__ __launch_bounds__(256) void pack_w_kernel(
    const float* __restrict__ w, _Float16* __restrict__ Bp, int N, int K) {
  int id = blockIdx.x * 256 + threadIdx.x;
  if (id >= N * K) return;
  int n = id / K;
  int k = id - n * K;
  Bp[((size_t)(k >> 5) * N + n) * 32 + (k & 31)] = (_Float16)w[id];
}

// =====================================================================
// im2col for conv1: input (512,25,25) fp32 -> A1 (544 x 4608) f16,
// rows >= 529 zero-filled (M padded to multiple of 16).
// =====================================================================
__global__ __launch_bounds__(256) void im2col1_kernel(
    const float* __restrict__ in, _Float16* __restrict__ A) {
  int id = blockIdx.x * 256 + threadIdx.x;
  if (id >= 544 * 4608) return;
  int m = id / 4608;
  int k = id - m * 4608;
  float v = 0.0f;
  if (m < 529) {
    int oh = m / 23, ow = m - oh * 23;
    int ci = k / 9, r = k - ci * 9;
    int kh = r / 3, kw = r - kh * 3;
    v = in[ci * 625 + (oh + kh) * 25 + (ow + kw)];
  }
  A[id] = (_Float16)v;
}

// =====================================================================
// MaxPool 2x2 stride 2 ceil-mode over relu'd conv1 output (512,23,23)
// -> pooled (512,12,12) f16 (window clipped at 23 == -inf padding).
// =====================================================================
__global__ __launch_bounds__(256) void pool_kernel(
    const float* __restrict__ act, _Float16* __restrict__ pooled) {
  int id = blockIdx.x * 256 + threadIdx.x;
  if (id >= 512 * 144) return;
  int c = id / 144, p = id - c * 144;
  int ph = p / 12, pw = p - ph * 12;
  int h0 = 2 * ph, w0 = 2 * pw;
  const float* base = act + (size_t)c * 529;
  float v = base[h0 * 23 + w0];
  if (w0 + 1 < 23) v = fmaxf(v, base[h0 * 23 + w0 + 1]);
  if (h0 + 1 < 23) {
    v = fmaxf(v, base[(h0 + 1) * 23 + w0]);
    if (w0 + 1 < 23) v = fmaxf(v, base[(h0 + 1) * 23 + w0 + 1]);
  }
  pooled[id] = (_Float16)v;
}

// =====================================================================
// im2col for conv2: pooled (512,12,12) f16 -> A2 (112 x 4608) f16.
// =====================================================================
__global__ __launch_bounds__(256) void im2col2_kernel(
    const _Float16* __restrict__ pooled, _Float16* __restrict__ A) {
  int id = blockIdx.x * 256 + threadIdx.x;
  if (id >= 112 * 4608) return;
  int m = id / 4608;
  int k = id - m * 4608;
  _Float16 v = (_Float16)0.0f;
  if (m < 100) {
    int oh = m / 10, ow = m - oh * 10;
    int ci = k / 9, r = k - ci * 9;
    int kh = r / 3, kw = r - kh * 3;
    v = pooled[ci * 144 + (oh + kh) * 12 + (ow + kw)];
  }
  A[id] = v;
}

// =====================================================================
// WMMA implicit-GEMM with fused bias + ReLU.
//   A  : M_pad x K f16 row-major (zero-padded rows)
//   Bp : [K/32][N][32] f16 packed column segments
//   out: N x M_real fp32 (channel-major == NCHW)
//
// Block = 4 waves sharing one 16-row M tile, covering 256 N columns
// (one 16x64 tile per wave, 4 f32 accumulators).
// A K-panels (16 x 1536 f16 = 48 KB) are staged into LDS with
// global_load_async_to_lds_b128 (ASYNCcnt) and read back as ds_load_b128.
// B fragments are double-buffered in registers so loads for k-block kb+1
// are in flight while the WMMAs for kb execute.
// =====================================================================
#define KPANEL 1536            // halves per LDS stage = 48 k-blocks
#define BSTEP_N 512            // halves between 16-wide N subtiles in Bp

__device__ __forceinline__ void load_b4(V16 b[4], const _Float16* bb) {
#pragma unroll
  for (int s = 0; s < 4; ++s) {
    b[s].h[0] = *(const v8h*)(bb + s * BSTEP_N);
    b[s].h[1] = *(const v8h*)(bb + s * BSTEP_N + 16);
  }
}

__device__ __forceinline__ void wmma4(v8f& a0, v8f& a1, v8f& a2, v8f& a3,
                                      const v16h av, const V16 b[4]) {
  a0 = __builtin_amdgcn_wmma_f32_16x16x32_f16(false, av, false, b[0].v,
                                              (short)0, a0, false, false);
  a1 = __builtin_amdgcn_wmma_f32_16x16x32_f16(false, av, false, b[1].v,
                                              (short)0, a1, false, false);
  a2 = __builtin_amdgcn_wmma_f32_16x16x32_f16(false, av, false, b[2].v,
                                              (short)0, a2, false, false);
  a3 = __builtin_amdgcn_wmma_f32_16x16x32_f16(false, av, false, b[3].v,
                                              (short)0, a3, false, false);
}

__global__ __launch_bounds__(128) void wmma_gemm_bias_relu_kernel(
    const _Float16* __restrict__ A, const _Float16* __restrict__ Bp,
    const float* __restrict__ bias, float* __restrict__ out,
    int M_real, int M_pad, int N, int K) {
  __shared__ _Float16 As[16 * KPANEL];   // 48 KB LDS panel

  const int tid  = threadIdx.x;
  const int lane = tid & 31;
  const int wave = tid >> 5;
  const int tiles_m = M_pad >> 4;
  const int tile_m  = blockIdx.x % tiles_m;
  const int nblk    = blockIdx.x / tiles_m;     // 256-wide N block
  const int g  = lane >> 4;                     // lane group
  const int ln = lane & 15;
  const int n0 = nblk * 256 + wave * 64 + ln;   // this lane's first N col

  const _Float16* Ag = A + (size_t)tile_m * 16 * K;
  const size_t bstep_k = (size_t)N * 32;        // halves per k-block in Bp
  const _Float16* Bbase = Bp + (size_t)n0 * 32 + 8 * g;

  v8f acc0 = {}, acc1 = {}, acc2 = {}, acc3 = {};

  const int nstage = K / KPANEL;                // 3
  for (int st = 0; st < nstage; ++st) {
    const int base_k = st * KPANEL;

    // ---- async-stage the 16 x KPANEL A panel into LDS ----
    __syncthreads();                            // prior-stage reads complete
#pragma unroll 1
    for (int c = tid; c < 16 * (KPANEL / 8); c += 128) {
      int row = c / (KPANEL / 8);
      int col = c - row * (KPANEL / 8);
      unsigned lds_off =
          (unsigned)(size_t)(const void*)&As[row * KPANEL + col * 8];
      unsigned long long ga =
          (unsigned long long)(size_t)(Ag + (size_t)row * K + base_k + col * 8);
      asm volatile("global_load_async_to_lds_b128 %0, %1, off"
                   :: "v"(lds_off), "v"(ga) : "memory");
      if (st + 1 < nstage)                      // warm next panel in GL2
        __builtin_prefetch(Ag + (size_t)row * K + base_k + KPANEL + col * 8, 0, 1);
    }
    asm volatile("s_wait_asynccnt 0x0" ::: "memory");
    __syncthreads();                            // panel visible to all waves

    const _Float16* arow = &As[ln * KPANEL + 8 * g];

    // ---- inner K loop over 48 k-blocks, double-buffered B ----
    const int kb0 = base_k >> 5;
    const int KBs = KPANEL / 32;                // 48 (even)
    V16 b0[4], b1[4];
    load_b4(b0, Bbase + (size_t)kb0 * bstep_k);
#pragma unroll 2
    for (int kb = 0; kb < KBs; kb += 2) {
      load_b4(b1, Bbase + (size_t)(kb0 + kb + 1) * bstep_k);
      V16 a0;
      a0.h[0] = *(const v8h*)(arow + (kb << 5));
      a0.h[1] = *(const v8h*)(arow + (kb << 5) + 16);
      wmma4(acc0, acc1, acc2, acc3, a0.v, b0);
      if (kb + 2 < KBs)
        load_b4(b0, Bbase + (size_t)(kb0 + kb + 2) * bstep_k);
      V16 a1;
      a1.h[0] = *(const v8h*)(arow + (kb << 5) + 32);
      a1.h[1] = *(const v8h*)(arow + (kb << 5) + 48);
      wmma4(acc0, acc1, acc2, acc3, a1.v, b1);
    }
  }

  // ---- epilogue: bias + relu, store channel-major (NCHW) ----
  const float bv0 = bias[n0];
  const float bv1 = bias[n0 + 16];
  const float bv2 = bias[n0 + 32];
  const float bv3 = bias[n0 + 48];
#pragma unroll
  for (int r = 0; r < 8; ++r) {
    const int mr = tile_m * 16 + 8 * g + r;
    if (mr < M_real) {
      out[(size_t)(n0)      * M_real + mr] = fmaxf(acc0[r] + bv0, 0.0f);
      out[(size_t)(n0 + 16) * M_real + mr] = fmaxf(acc1[r] + bv1, 0.0f);
      out[(size_t)(n0 + 32) * M_real + mr] = fmaxf(acc2[r] + bv2, 0.0f);
      out[(size_t)(n0 + 48) * M_real + mr] = fmaxf(acc3[r] + bv3, 0.0f);
    }
  }
}

// =====================================================================
// Gram: x flat (102400 fp32) viewed as (100,1024); gram[l,i,j]=x[l,i]*x[l,j].
// One thread per 4 output floats; 419 MB of non-temporal b128 stores is the
// pipeline's roofline limiter (~18 us at 23.3 TB/s HBM).
// =====================================================================
__global__ __launch_bounds__(256) void gram_kernel(
    const float* __restrict__ x, float* __restrict__ out) {
  unsigned t   = blockIdx.x * 256u + threadIdx.x;   // < 26,214,400
  unsigned j4  = t & 255u;
  unsigned rem = t >> 8;
  unsigned i   = rem & 1023u;
  unsigned l   = rem >> 10;
  const float xi = x[(l << 10) + i];
  const v4f  xj = *(const v4f*)(x + (l << 10) + (j4 << 2));
  v4f r;
  r.x = xi * xj.x; r.y = xi * xj.y; r.z = xi * xj.z; r.w = xi * xj.w;
  __builtin_nontemporal_store(r, (v4f*)out + t);
}

// =====================================================================
// Host launch
// =====================================================================
extern "C" void kernel_launch(void* const* d_in, const int* in_sizes, int n_in,
                              void* d_out, int out_size, void* d_ws, size_t ws_size,
                              hipStream_t stream) {
  const float* input = (const float*)d_in[0];   // (1,512,25,25)
  const float* w1    = (const float*)d_in[1];   // (512,512,3,3)
  const float* b1    = (const float*)d_in[2];   // (512)
  const float* w2    = (const float*)d_in[3];   // (1024,512,3,3)
  const float* b2    = (const float*)d_in[4];   // (1024)
  float* out = (float*)d_out;                   // (100,1024,1024)

  char* ws = (char*)d_ws;
  _Float16* w1p    = (_Float16*)(ws);                         //  4,718,592 B
  _Float16* w2p    = (_Float16*)(ws + 4718592);               //  9,437,184 B
  _Float16* A1     = (_Float16*)(ws + 14155776);              //  5,013,504 B
  float*    act1   = (float*)   (ws + 19169280);              //  1,083,392 B
  _Float16* pooled = (_Float16*)(ws + 20252672);              //    147,456 B
  _Float16* A2     = (_Float16*)(ws + 20400128);              //  1,032,192 B
  float*    out2   = (float*)   (ws + 21432320);              //    409,600 B

  // 1) pack weights into B-fragment layout
  pack_w_kernel<<<(512 * 4608 + 255) / 256, 256, 0, stream>>>(w1, w1p, 512, 4608);
  pack_w_kernel<<<(1024 * 4608 + 255) / 256, 256, 0, stream>>>(w2, w2p, 1024, 4608);

  // 2) im2col conv1 operand (544 x 4608)
  im2col1_kernel<<<(544 * 4608) / 256, 256, 0, stream>>>(input, A1);

  // 3) conv1 GEMM: M=529(pad 544), N=512, K=4608 -> act1 (512 x 529), relu
  {
    int blocks = (544 / 16) * (512 / 256);      // 34 * 2 = 68
    wmma_gemm_bias_relu_kernel<<<blocks, 128, 0, stream>>>(
        A1, w1p, b1, act1, 529, 544, 512, 4608);
  }

  // 4) maxpool 2x2 ceil -> pooled f16 (512 x 144)
  pool_kernel<<<(512 * 144) / 256, 256, 0, stream>>>(act1, pooled);

  // 5) im2col conv2 operand (112 x 4608)
  im2col2_kernel<<<(112 * 4608) / 256, 256, 0, stream>>>(pooled, A2);

  // 6) conv2 GEMM: M=100(pad 112), N=1024, K=4608 -> out2 (1024 x 100), relu
  {
    int blocks = (112 / 16) * (1024 / 256);     // 7 * 4 = 28
    wmma_gemm_bias_relu_kernel<<<blocks, 128, 0, stream>>>(
        A2, w2p, b2, out2, 100, 112, 1024, 4608);
  }

  // 7) Gram outer products, streaming non-temporal stores (419 MB)
  gram_kernel<<<(100u * 1024u * 1024u / 4u) / 256u, 256, 0, stream>>>(out2, out);
}
